// NonlinearEncoder_19447611916913
// MI455X (gfx1250) — matrix operand order repaired and checked by reading, not compiled, not run on hardware
//
#include <hip/hip_runtime.h>
#include <hip/hip_bf16.h>

// ---------------------------------------------------------------------------
// Fused grouped-MLP encoder for MI455X (gfx1250, wave32, WMMA).
//   mu[b,j] = W1[j] . leakyrelu( (x[b,:]*w[i,j,:]) @ W0[j]^T + b0[j] ) + b1[j]
// One workgroup = (128 rows of B) x (one j). bf16 WMMA 16x16x32, f32 accum.
// Double-buffered W0 staging (global loads in flight during WMMA chain),
// all fragments preloaded before the WMMA chain, shuffle-reduced epilogue.
// ---------------------------------------------------------------------------

typedef __attribute__((ext_vector_type(16))) __bf16 v16bf;
typedef __attribute__((ext_vector_type(8)))  float  v8f;

#define D_  4
#define J_  256
#define K_  512
#define H_  512
#define B_  1024

#define BM  128            // rows of x per workgroup
#define BN  64             // columns of H per n-tile
#define KC  64             // k-chunk staged per step
#define NKC (K_ / KC)      // 8
#define XSTR (K_ + 8)      // 520 bf16 -> 1040B row stride (16B aligned, conflict-free)
#define WSTR (BN + 8)      // 72  bf16 -> 144B  row stride (16B aligned, conflict-free)

#define XS_BYTES (BM * XSTR * 2)   // 133120
#define WS_BYTES (KC * WSTR * 2)   // 9216 per buffer
#define SMEM_BYTES (XS_BYTES + 2*WS_BYTES + K_*4 + H_*4 + H_*4 + BM*4)  // 158208

__device__ __forceinline__ unsigned short f32_to_bf16(float f) {
  unsigned int u = __float_as_uint(f);
  unsigned int r = 0x7FFFu + ((u >> 16) & 1u);   // round-to-nearest-even
  return (unsigned short)((u + r) >> 16);
}

union Frag { v16bf v; uint4 u[2]; };

// Fetch one 64x64 f32 chunk of W0[j, n0:n0+64, kc*64 : kc*64+64] -> 4 float4/thread.
__device__ __forceinline__ void w0_fetch(float4* g, const float* __restrict__ W0,
                                         size_t rowbase, int kcc, int tid) {
  #pragma unroll
  for (int q = 0; q < 4; ++q) {
    const int f  = tid + q * 256;      // float4 index in [0,1024)
    const int n  = f >> 4;             // row 0..63
    const int k0 = (f & 15) * 4;       // k within chunk
    g[q] = *(const float4*)&W0[(rowbase + n) * K_ + kcc * KC + k0];
  }
}

// Convert + scatter into B-matrix (K x N) bf16 layout: Wd[k][n].
__device__ __forceinline__ void w0_store(unsigned short* Wd, const float4* g, int tid) {
  #pragma unroll
  for (int q = 0; q < 4; ++q) {
    const int f  = tid + q * 256;
    const int n  = f >> 4;
    const int k0 = (f & 15) * 4;
    Wd[(k0 + 0) * WSTR + n] = f32_to_bf16(g[q].x);
    Wd[(k0 + 1) * WSTR + n] = f32_to_bf16(g[q].y);
    Wd[(k0 + 2) * WSTR + n] = f32_to_bf16(g[q].z);
    Wd[(k0 + 3) * WSTR + n] = f32_to_bf16(g[q].w);
  }
}

extern "C" __global__ __launch_bounds__(256)
void nle_fused_wmma(const float* __restrict__ x,  const int* __restrict__ ip,
                    const float* __restrict__ w,  const float* __restrict__ W0,
                    const float* __restrict__ b0, const float* __restrict__ W1,
                    const float* __restrict__ b1, const float* __restrict__ logvar,
                    float* __restrict__ out)
{
  extern __shared__ unsigned char smem[];
  unsigned short* Xs  = (unsigned short*)smem;                       // [BM][XSTR] bf16
  unsigned short* Ws  = (unsigned short*)(smem + XS_BYTES);          // 2 x [KC][WSTR] bf16
  float*          wiS = (float*)(smem + XS_BYTES + 2*WS_BYTES);      // [K]
  float*          w1S = wiS + K_;                                    // [H]
  float*          b0S = w1S + H_;                                    // [H]
  float*          muS = b0S + H_;                                    // [BM]

  const int tid = threadIdx.x;
  const int j   = blockIdx.y;
  const int m0  = blockIdx.x * BM;
  const int i   = ip[0];
  const size_t w0row = (size_t)j * H_;    // row base of W0[j]

  // ---- stage per-j vectors: mask row, W1 row, b0 row; zero mu accumulator ----
  const float* wrow = w + ((size_t)i * J_ + j) * (size_t)K_;
  for (int t = tid; t < K_; t += 256) wiS[t] = wrow[t];
  for (int t = tid; t < H_; t += 256) {
    w1S[t] = W1[w0row + t];
    b0S[t] = b0[w0row + t];
  }
  if (tid < BM) muS[tid] = 0.0f;
  __syncthreads();

  // ---- stage full A-tile once: Xs[m][k] = bf16( x[m0+m,k] * wi[k] ) ----
  for (int idx = tid; idx < BM * K_; idx += 256) {
    const int m = idx >> 9;          // /K_
    const int k = idx & (K_ - 1);
    const float v = x[(size_t)(m0 + m) * K_ + k] * wiS[k];
    Xs[m * XSTR + k] = f32_to_bf16(v);
  }
  __syncthreads();

  const int wave  = tid >> 5;               // 8 waves, each owns 16 M-rows
  const int lane  = tid & 31;
  const int arow  = wave * 16 + (lane & 15);
  const int ahalf = (lane >> 4) << 3;       // 0 for lanes 0-15, 8 for lanes 16-31

  v8f acc[4];

  for (int nt = 0; nt < H_ / BN; ++nt) {
    const int n0 = nt * BN;
    #pragma unroll
    for (int ns = 0; ns < 4; ++ns)
      #pragma unroll
      for (int v = 0; v < 8; ++v) acc[ns][v] = 0.0f;

    // prologue: stage chunk 0 into buffer 0
    {
      float4 g[4];
      w0_fetch(g, W0, w0row + n0, 0, tid);
      w0_store(Ws, g, tid);
    }
    __syncthreads();

    for (int kc = 0; kc < NKC; ++kc) {
      const int p = kc & 1;
      const unsigned short* Wcur = Ws + p * (KC * WSTR);
      unsigned short*       Wnxt = Ws + (p ^ 1) * (KC * WSTR);

      // Phase 1: issue next chunk's global loads (latency hidden by WMMAs)
      float4 g[4];
      if (kc + 1 < NKC) w0_fetch(g, W0, w0row + n0, kc + 1, tid);
      if (kc + 2 < NKC)
        __builtin_prefetch(&W0[(w0row + n0 + (tid & (BN - 1))) * K_ + (kc + 2) * KC], 0, 0);

      // Phase 2: preload ALL fragments, then run the WMMA chain
      Frag A[2], Bf[2][4];
      #pragma unroll
      for (int ks = 0; ks < 2; ++ks) {
        const int abase = arow * XSTR + kc * KC + ks * 32 + ahalf;
        A[ks].u[0] = *(const uint4*)(Xs + abase);
        A[ks].u[1] = *(const uint4*)(Xs + abase + 16);
        const int brow = ks * 32 + ahalf;    // B layout: lane holds column lane&15, K rows ahalf+0..7 and 16+ahalf+0..7
        #pragma unroll
        for (int ns = 0; ns < 4; ++ns) {
          const int bbase = brow * WSTR + ns * 16 + (lane & 15);
          #pragma unroll
          for (int s = 0; s < 16; ++s)
            Bf[ks][ns].v[s] = __builtin_bit_cast(__bf16, Wcur[bbase + ((s & 7) + (s >> 3) * 16) * WSTR]);
        }
      }
      #pragma unroll
      for (int ks = 0; ks < 2; ++ks)
        #pragma unroll
        for (int ns = 0; ns < 4; ++ns)
          acc[ns] = __builtin_amdgcn_wmma_f32_16x16x32_bf16(
              false, A[ks].v, false, Bf[ks][ns].v, (short)0, acc[ns], false, false);

      // Phase 3: convert + store next chunk into the other buffer
      if (kc + 1 < NKC) w0_store(Wnxt, g, tid);
      __syncthreads();
    }

    // ---- epilogue: +b0, LeakyReLU(0.01), dot with W1 row, butterfly-reduce ----
    // C/D layout: lane's column n = lane&15; VGPR v -> row = 8*(lane>=16)+v.
    float part[8];
    #pragma unroll
    for (int v = 0; v < 8; ++v) part[v] = 0.0f;
    #pragma unroll
    for (int ns = 0; ns < 4; ++ns) {
      const int   ncol = n0 + ns * 16 + (lane & 15);
      const float w1v  = w1S[ncol];
      const float b0v  = b0S[ncol];
      #pragma unroll
      for (int v = 0; v < 8; ++v) {
        float h = acc[ns][v] + b0v;
        h = h > 0.0f ? h : 0.01f * h;
        part[v] += h * w1v;
      }
    }
    // reduce over the 16 columns; masks 1..8 stay inside each 16-lane half
    #pragma unroll
    for (int msk = 1; msk < 16; msk <<= 1)
      #pragma unroll
      for (int v = 0; v < 8; ++v)
        part[v] += __shfl_xor(part[v], msk, 32);
    if ((lane & 15) == 0) {     // lane 0 -> rows 0-7 of block; lane 16 -> rows 8-15
      #pragma unroll
      for (int v = 0; v < 8; ++v)
        muS[wave * 16 + ahalf + v] += part[v];   // exclusive rows: no atomic needed
    }
  }
  __syncthreads();

  if (tid < BM)
    out[(size_t)(m0 + tid) * J_ + j] = muS[tid] + b1[j];

  // second tuple element: logvar (4 floats) appended after mu
  if (blockIdx.x == 0 && blockIdx.y == 0 && tid < D_)
    out[(size_t)B_ * J_ + tid] = logvar[tid];
}

extern "C" void kernel_launch(void* const* d_in, const int* in_sizes, int n_in,
                              void* d_out, int out_size, void* d_ws, size_t ws_size,
                              hipStream_t stream) {
  (void)in_sizes; (void)n_in; (void)d_ws; (void)ws_size; (void)out_size;
  const float* x      = (const float*)d_in[0];
  const int*   ip     = (const int*)  d_in[1];
  const float* w      = (const float*)d_in[2];
  const float* W0     = (const float*)d_in[3];
  const float* b0     = (const float*)d_in[4];
  const float* W1     = (const float*)d_in[5];
  const float* b1     = (const float*)d_in[6];
  const float* logvar = (const float*)d_in[7];
  float* out = (float*)d_out;

  // ~158 KB dynamic LDS per workgroup (CDNA5 WGP has 320 KB) -> raise the cap.
  (void)hipFuncSetAttribute((const void*)nle_fused_wmma,
                            hipFuncAttributeMaxDynamicSharedMemorySize, SMEM_BYTES);

  dim3 grid(B_ / BM, J_);   // x fastest -> same-j workgroups adjacent (L2 reuse of W0[j])
  nle_fused_wmma<<<grid, 256, SMEM_BYTES, stream>>>(x, ip, w, W0, b0, W1, b1, logvar, out);
}
